// GaussianVectorQuantizer_5669356831648
// MI455X (gfx1250) — matrix operand order repaired and checked by reading, not compiled, not run on hardware
//
#include <hip/hip_runtime.h>
#include <hip/hip_bf16.h>

// ---------------------------------------------------------------------------
// Gaussian Vector Quantizer (fused) for MI455X / gfx1250, wave32 + WMMA bf16.
//   N = 32768 rows (bs*W*H), D = 256 (dim_z), K = 1024 (size_dict)
//   GEMM1: logits = -(w)*(||z||^2 + ||c||^2 - 2 z@C^T)      [N,K]
//   GEMM2: z_q   = softmax((logits+g)/T) @ C                [N,D]
// Both GEMMs run on v_wmma_f32_16x16x32_bf16 with f32 accumulation.
// ---------------------------------------------------------------------------

typedef __attribute__((ext_vector_type(16))) __bf16 v16bf;
typedef __attribute__((ext_vector_type(8)))  float  v8f;

#define DIMZ   256
#define KDICT  1024
#define NROWS  32768
#define BSZ    32
#define TEMP_INV 2.0f      // 1/0.5
#define EPS_G  1e-10f
#define EPS_V  1e-10f
#define EPS_P  1e-7f

// workspace layout (bytes)
#define WS_CBB   0u          // bf16 codebook [K][D]           (512 KB)
#define WS_CBT   524288u     // bf16 codebook^T [D][K]         (512 KB)
#define WS_CNORM 1048576u    // f32  ||c_k||^2 [K]             (4 KB)
#define WS_ACCP  1052672u    // f32  column sums of p [K]      (4 KB)
#define WS_ACC   1056768u    // f32  {sum p*logp, sum diff^2}  (8 B)

// ---------------------------------------------------------------------------
__global__ __launch_bounds__(256) void vq_init(const float* __restrict__ cb,
                                               __bf16* __restrict__ cbB,
                                               __bf16* __restrict__ cbT,
                                               float* __restrict__ cnorm,
                                               float* __restrict__ accP,
                                               float* __restrict__ accS) {
  const int k = blockIdx.x;      // codebook entry
  const int t = threadIdx.x;     // dim index (256)
  float v = cb[k * DIMZ + t];
  cbB[k * DIMZ + t]  = (__bf16)v;
  cbT[t * KDICT + k] = (__bf16)v;
  float p = v * v;
  #pragma unroll
  for (int s = 16; s > 0; s >>= 1) p += __shfl_xor(p, s, 32);
  __shared__ float red[8];
  if ((t & 31) == 0) red[t >> 5] = p;
  __syncthreads();
  if (t == 0) {
    float s = 0.f;
    #pragma unroll
    for (int i = 0; i < 8; ++i) s += red[i];
    cnorm[k] = s;
  }
  // zero global accumulators (re-done every launch: deterministic)
  if (blockIdx.x < 4) accP[blockIdx.x * 256 + t] = 0.f;
  if (blockIdx.x == 0 && t < 2) accS[t] = 0.f;
}

// ---------------------------------------------------------------------------
__global__ __launch_bounds__(256) void vq_main(const float* __restrict__ zf,
                                               const float* __restrict__ varq,
                                               const float* __restrict__ gu,
                                               const __bf16* __restrict__ cbB,
                                               const __bf16* __restrict__ cbT,
                                               const float* __restrict__ cnorm,
                                               float* __restrict__ accP,
                                               float* __restrict__ accS,
                                               float* __restrict__ out) {
  __shared__ float  zLDS[16 * DIMZ];     // fp32 z tile; later reused for z_q
  __shared__ __bf16 encLDS[16 * KDICT];  // bf16 encodings (GEMM2 A)
  __shared__ float  cnLDS[KDICT];        // codebook norms
  __shared__ float  redLDS[16 * 8];      // row-stat scratch [m][wave]
  __shared__ float  znLDS[16];           // row norms of z

  const int t    = threadIdx.x;
  const int wave = t >> 5;
  const int lane = t & 31;
  const int l16  = lane & 15;
  const int half = lane >> 4;
  const int rowbase = blockIdx.x * 16;

  const float weight = 0.5f / fmaxf(varq[0], EPS_V);

  // ---- phase 0: load z tile (coalesced: 16 consecutive h per d) -----------
  if (t < 16) znLDS[t] = 0.f;
  __syncthreads();
  {
    const int  nl   = t & 15;
    const int  n    = rowbase + nl;                 // global row
    const long base = (long)(n >> 10) * (DIMZ * 1024) + (n & 1023);
    float zn = 0.f;
    #pragma unroll
    for (int p = 0; p < 16; ++p) {
      int d = p * 16 + (t >> 4);
      float v = zf[base + (long)d * 1024];
      zLDS[nl * DIMZ + d] = v;
      zn += v * v;
    }
    atomicAdd(&znLDS[nl], zn);
    #pragma unroll
    for (int i = 0; i < 4; ++i) cnLDS[t + i * 256] = cnorm[t + i * 256];
  }
  __syncthreads();

  // ---- phase 1: GEMM1  logits-tile in registers ----------------------------
  // wave owns codebook entries [wave*128, wave*128+128): 8 tiles of 16
  v8f acc[8];
  #pragma unroll
  for (int i = 0; i < 8; ++i) acc[i] = v8f{0.f,0.f,0.f,0.f,0.f,0.f,0.f,0.f};

  #pragma unroll
  for (int kt = 0; kt < 8; ++kt) {
    const int kbase = kt * 32;
    v16bf a;                                        // A: 16x32 bf16 (z rows)
    #pragma unroll
    for (int v = 0; v < 8; ++v) {
      int k0 = kbase + half * 16 + 2 * v;
      a[2 * v]     = (__bf16)zLDS[l16 * DIMZ + k0];
      a[2 * v + 1] = (__bf16)zLDS[l16 * DIMZ + k0 + 1];
    }
    #pragma unroll
    for (int nt = 0; nt < 8; ++nt) {
      const int entry = wave * 128 + nt * 16 + l16;
      const __bf16* bp = cbB + (size_t)entry * DIMZ + kbase + half * 16;
      __builtin_prefetch(bp + 32, 0, 0);            // global_prefetch_b8
      v16bf b;                                      // B: 32x16, K = half*16+2v
      #pragma unroll
      for (int v = 0; v < 8; ++v) { b[2 * v] = bp[2 * v]; b[2 * v + 1] = bp[2 * v + 1]; }
      acc[nt] = __builtin_amdgcn_wmma_f32_16x16x32_bf16(
          false, a, false, b, (short)0, acc[nt], false, false);
    }
  }

  // ---- phase 2: dot -> logits ---------------------------------------------
  #pragma unroll
  for (int nt = 0; nt < 8; ++nt) {
    const float cn = cnLDS[wave * 128 + nt * 16 + l16];
    #pragma unroll
    for (int vv = 0; vv < 8; ++vv) {
      const float zn = znLDS[vv + 8 * half];        // C layout: m = vv + 8*half
      acc[nt][vv] = weight * (2.f * acc[nt][vv] - zn - cn);
    }
  }

  // row-wide reductions over k (16 lanes of a half × 8 tiles × 8 waves)
  auto rowMax = [&](float (&loc)[8]) {
    #pragma unroll
    for (int vv = 0; vv < 8; ++vv) {
      float x = loc[vv];
      x = fmaxf(x, __shfl_xor(x, 1, 32));
      x = fmaxf(x, __shfl_xor(x, 2, 32));
      x = fmaxf(x, __shfl_xor(x, 4, 32));
      x = fmaxf(x, __shfl_xor(x, 8, 32));
      loc[vv] = x;
    }
    __syncthreads();
    if (l16 == 0) {
      #pragma unroll
      for (int vv = 0; vv < 8; ++vv) redLDS[(vv + 8 * half) * 8 + wave] = loc[vv];
    }
    __syncthreads();
    #pragma unroll
    for (int vv = 0; vv < 8; ++vv) {
      const int m = vv + 8 * half;
      float x = redLDS[m * 8];
      #pragma unroll
      for (int w2 = 1; w2 < 8; ++w2) x = fmaxf(x, redLDS[m * 8 + w2]);
      loc[vv] = x;
    }
  };
  auto rowSum = [&](float (&loc)[8]) {
    #pragma unroll
    for (int vv = 0; vv < 8; ++vv) {
      float x = loc[vv];
      x += __shfl_xor(x, 1, 32);
      x += __shfl_xor(x, 2, 32);
      x += __shfl_xor(x, 4, 32);
      x += __shfl_xor(x, 8, 32);
      loc[vv] = x;
    }
    __syncthreads();
    if (l16 == 0) {
      #pragma unroll
      for (int vv = 0; vv < 8; ++vv) redLDS[(vv + 8 * half) * 8 + wave] = loc[vv];
    }
    __syncthreads();
    #pragma unroll
    for (int vv = 0; vv < 8; ++vv) {
      const int m = vv + 8 * half;
      float x = 0.f;
      #pragma unroll
      for (int w2 = 0; w2 < 8; ++w2) x += redLDS[m * 8 + w2];
      loc[vv] = x;
    }
  };

  float rm[8], rs[8];
  #pragma unroll
  for (int vv = 0; vv < 8; ++vv) {
    float x = -3.4e38f;
    #pragma unroll
    for (int nt = 0; nt < 8; ++nt) x = fmaxf(x, acc[nt][vv]);
    rm[vv] = x;
  }
  rowMax(rm);
  #pragma unroll
  for (int vv = 0; vv < 8; ++vv) {
    float s = 0.f;
    #pragma unroll
    for (int nt = 0; nt < 8; ++nt) s += __expf(acc[nt][vv] - rm[vv]);
    rs[vv] = s;
  }
  rowSum(rs);

  // ---- phase 3: p stats: sum p*log p, column sums for avg_probs -----------
  float lrs[8], inv[8];
  #pragma unroll
  for (int vv = 0; vv < 8; ++vv) { lrs[vv] = __logf(rs[vv]); inv[vv] = 1.f / rs[vv]; }
  float plogp = 0.f;
  #pragma unroll
  for (int nt = 0; nt < 8; ++nt) {
    float cs = 0.f;
    #pragma unroll
    for (int vv = 0; vv < 8; ++vv) {
      const float sh = acc[nt][vv] - rm[vv];
      const float p  = __expf(sh) * inv[vv];
      plogp += p * (sh - lrs[vv]);
      cs += p;
    }
    cs += __shfl_xor(cs, 16, 32);                   // combine both m-halves
    if (half == 0) atomicAdd(&accP[wave * 128 + nt * 16 + l16], cs);
  }
  #pragma unroll
  for (int s = 16; s > 0; s >>= 1) plogp += __shfl_xor(plogp, s, 32);
  if (lane == 0) atomicAdd(&accS[0], plogp);

  // ---- phase 4: gumbel-softmax encodings -> LDS (bf16) ---------------------
  #pragma unroll
  for (int nt = 0; nt < 8; ++nt) {
    const int k = wave * 128 + nt * 16 + l16;
    #pragma unroll
    for (int vv = 0; vv < 8; ++vv) {
      const int n = rowbase + vv + 8 * half;
      const float u = gu[(size_t)n * KDICT + k];
      const float g = -__logf(-__logf(u + EPS_G) + EPS_G);
      acc[nt][vv] = (acc[nt][vv] + g) * TEMP_INV;
    }
  }
  #pragma unroll
  for (int vv = 0; vv < 8; ++vv) {
    float x = -3.4e38f;
    #pragma unroll
    for (int nt = 0; nt < 8; ++nt) x = fmaxf(x, acc[nt][vv]);
    rm[vv] = x;
  }
  rowMax(rm);
  #pragma unroll
  for (int vv = 0; vv < 8; ++vv) {
    float s = 0.f;
    #pragma unroll
    for (int nt = 0; nt < 8; ++nt) s += __expf(acc[nt][vv] - rm[vv]);
    rs[vv] = s;
  }
  rowSum(rs);
  #pragma unroll
  for (int vv = 0; vv < 8; ++vv) inv[vv] = 1.f / rs[vv];
  #pragma unroll
  for (int nt = 0; nt < 8; ++nt) {
    const int k = wave * 128 + nt * 16 + l16;
    #pragma unroll
    for (int vv = 0; vv < 8; ++vv) {
      const int m = vv + 8 * half;
      const float e = __expf(acc[nt][vv] - rm[vv]) * inv[vv];
      encLDS[m * KDICT + k] = (__bf16)e;
    }
  }
  __syncthreads();

  // ---- phase 5: GEMM2  z_q = enc @ C  (wave owns d in [wave*32, +32)) -----
  v8f accq[2];
  #pragma unroll
  for (int i = 0; i < 2; ++i) accq[i] = v8f{0.f,0.f,0.f,0.f,0.f,0.f,0.f,0.f};
  #pragma unroll 4
  for (int kt = 0; kt < 32; ++kt) {
    const int kbase = kt * 32;
    v16bf a;
    #pragma unroll
    for (int v = 0; v < 8; ++v) {
      int k0 = kbase + half * 16 + 2 * v;
      a[2 * v]     = encLDS[l16 * KDICT + k0];
      a[2 * v + 1] = encLDS[l16 * KDICT + k0 + 1];
    }
    #pragma unroll
    for (int dt = 0; dt < 2; ++dt) {
      const int d = wave * 32 + dt * 16 + l16;
      const __bf16* bp = cbT + (size_t)d * KDICT + kbase + half * 16;
      __builtin_prefetch(bp + 32, 0, 0);
      v16bf b;
      #pragma unroll
      for (int v = 0; v < 8; ++v) { b[2 * v] = bp[2 * v]; b[2 * v + 1] = bp[2 * v + 1]; }
      accq[dt] = __builtin_amdgcn_wmma_f32_16x16x32_bf16(
          false, a, false, b, (short)0, accq[dt], false, false);
    }
  }

  // ---- phase 6: kld_continuous partial + z_q into LDS (in place) ----------
  float c2 = 0.f;
  #pragma unroll
  for (int dt = 0; dt < 2; ++dt) {
    const int d = wave * 32 + dt * 16 + l16;
    #pragma unroll
    for (int vv = 0; vv < 8; ++vv) {
      const int m = vv + 8 * half;
      const float zq = accq[dt][vv];
      const float df = zLDS[m * DIMZ + d] - zq;
      c2 += df * df;
      zLDS[m * DIMZ + d] = zq;                      // owner-exclusive slot
    }
  }
  #pragma unroll
  for (int s = 16; s > 0; s >>= 1) c2 += __shfl_xor(c2, s, 32);
  if (lane == 0) atomicAdd(&accS[1], c2);
  __syncthreads();

  // ---- phase 7: transposed, coalesced writeback ----------------------------
  {
    const int  nl   = t & 15;
    const int  n    = rowbase + nl;
    const long base = (long)(n >> 10) * (DIMZ * 1024) + (n & 1023);
    #pragma unroll
    for (int p = 0; p < 16; ++p) {
      int d = p * 16 + (t >> 4);
      out[base + (long)d * 1024] = zLDS[nl * DIMZ + d];
    }
  }
}

// ---------------------------------------------------------------------------
__global__ __launch_bounds__(256) void vq_final(const float* __restrict__ varq,
                                                const float* __restrict__ accP,
                                                const float* __restrict__ accS,
                                                float* __restrict__ out,
                                                int out_size) {
  const int t = threadIdx.x;
  float s = 0.f;
  #pragma unroll
  for (int i = t; i < KDICT; i += 256) {
    const float avg = accP[i] * (1.0f / (float)NROWS);
    s += avg * __logf(avg + EPS_P);
  }
  #pragma unroll
  for (int k = 16; k > 0; k >>= 1) s += __shfl_xor(s, k, 32);
  __shared__ float red[8];
  if ((t & 31) == 0) red[t >> 5] = s;
  __syncthreads();
  if (t == 0) {
    float tot = 0.f;
    #pragma unroll
    for (int i = 0; i < 8; ++i) tot += red[i];
    const float weight = 0.5f / fmaxf(varq[0], EPS_V);
    const float loss = accS[0] / (float)BSZ + weight * accS[1] / (float)BSZ;
    out[out_size - 2] = loss;
    out[out_size - 1] = __expf(-tot);
  }
}

// ---------------------------------------------------------------------------
extern "C" void kernel_launch(void* const* d_in, const int* in_sizes, int n_in,
                              void* d_out, int out_size, void* d_ws, size_t ws_size,
                              hipStream_t stream) {
  const float* zf   = (const float*)d_in[0];  // [32,256,32,32]
  const float* varq = (const float*)d_in[1];  // scalar
  const float* cb   = (const float*)d_in[2];  // [1024,256]
  const float* gu   = (const float*)d_in[3];  // [32768,1024]
  float* out = (float*)d_out;

  char* ws = (char*)d_ws;
  __bf16* cbB  = (__bf16*)(ws + WS_CBB);
  __bf16* cbT  = (__bf16*)(ws + WS_CBT);
  float* cnorm = (float*)(ws + WS_CNORM);
  float* accP  = (float*)(ws + WS_ACCP);
  float* accS  = (float*)(ws + WS_ACC);

  vq_init<<<KDICT, DIMZ, 0, stream>>>(cb, cbB, cbT, cnorm, accP, accS);
  vq_main<<<NROWS / 16, 256, 0, stream>>>(zf, varq, gu, cbB, cbT, cnorm, accP, accS, out);
  vq_final<<<1, 256, 0, stream>>>(varq, accP, accS, out, out_size);
}
